// MobiusGRU_17635135717379
// MI455X (gfx1250) — matrix operand order repaired and checked by reading, not compiled, not run on hardware
//
#include <hip/hip_runtime.h>
#include <cmath>

typedef __attribute__((ext_vector_type(16))) _Float16 v16h;
typedef __attribute__((ext_vector_type(8)))  float    v8f;
typedef __attribute__((ext_vector_type(4)))  int      v4i;

#if __has_builtin(__builtin_amdgcn_global_load_async_to_lds_b128) && \
    __has_builtin(__builtin_amdgcn_s_wait_asynccnt)
#define HAVE_ASYNC 1
#else
#define HAVE_ASYNC 0
#endif

#define AS_GLOBAL __attribute__((address_space(1)))
#define AS_LDS    __attribute__((address_space(3)))

namespace {
constexpr int   T    = 512;
constexpr int   NB   = 128;           // batch
constexpr int   H    = 256;           // hidden == input
constexpr int   TB   = T * NB;        // 65536 rows
constexpr long  TBH  = (long)TB * H;  // elements per gate slab
constexpr float EPSF = 1e-5f;
}

// ---------------- common device helpers ----------------

__device__ __forceinline__ float wave_sum(float v) {
#pragma unroll
  for (int m = 16; m >= 1; m >>= 1) v += __shfl_xor(v, m, 32);
  return v;
}

__device__ __forceinline__ float artanh_f(float x) {
  x = fminf(fmaxf(x, -1.0f + EPSF), 1.0f - EPSF);
  return 0.5f * (log1pf(x) - log1pf(-x));
}

__device__ __forceinline__ float clamp_norm(float sumsq) {
  return fmaxf(sqrtf(fmaxf(sumsq, 1e-30f)), EPSF);
}

// A fragment (16x32 f16, row-major source, leading dim ld), k = 16*(lane/16) + element, not the ISA table order:
// lanes 0-15: rows 0-15, VGPR v -> K=2v,2v+1 (K 0-15)
// lanes 16-31: same rows, K 16-31 (contiguous, matches B fragment).
__device__ __forceinline__ v16h load_a_frag(const _Float16* __restrict__ src,
                                            int ld, int k0, int lane) {
  int row = lane & 15;
  int ko  = (lane & 16) ? 16 : 0;
  v16h a;
#pragma unroll
  for (int v = 0; v < 8; ++v) {
    int kb = k0 + ko + 2 * v;
    a[2 * v]     = src[row * ld + kb];
    a[2 * v + 1] = src[row * ld + kb + 1];
  }
  return a;
}

// B fragment (32x16) from row-major f16 weight matrix Wg[H][H] as B = Wg^T:
// B[k][n] = Wg[n][k]; per lane 16 contiguous halves -> one aligned 32B load.
__device__ __forceinline__ v16h load_b_frag_h(const _Float16* __restrict__ wg,
                                              int ldw, int nbase, int k0, int lane) {
  int col = nbase + (lane & 15);
  int kb  = k0 + ((lane & 16) ? 16 : 0);
  return *(const v16h*)(wg + (long)col * ldw + kb);
}

// C/D 16x16 f32: VGPR r -> row r (lanes 0-15) / row r+8 (lanes 16-31), col = lane%16.
__device__ __forceinline__ void store_c(float* __restrict__ buf, int ld, int nbase,
                                        v8f c, int lane) {
  int col   = nbase + (lane & 15);
  int rbase = (lane & 16) ? 8 : 0;
#pragma unroll
  for (int r = 0; r < 8; ++r) buf[(rbase + r) * ld + col] = c[r];
}

// ---------------- kernel 0: f32 -> f16 weight conversion (one-time) ----------
__global__ void __launch_bounds__(256)
wcvt_kernel(const float* __restrict__ w, _Float16* __restrict__ o, int n) {
  int i = blockIdx.x * 256 + threadIdx.x;
  if (i < n) o[i] = (_Float16)w[i];
}

// ---------------- kernel 1: expmap0 + f16 + row norms ----------------
// block = 256 threads = 8 waves, one wave per row of H=256 elements.
__global__ void __launch_bounds__(256)
prep_kernel(const float* __restrict__ in, _Float16* __restrict__ outh,
            float* __restrict__ outn, int do_expmap) {
  int wv = threadIdx.x >> 5, lane = threadIdx.x & 31;
  long row = (long)blockIdx.x * 8 + wv;
  const float* r = in + row * H;
  float u[8], s = 0.f;
#pragma unroll
  for (int i = 0; i < 8; ++i) { u[i] = r[lane * 8 + i]; s += u[i] * u[i]; }
  s = wave_sum(s);
  float n = clamp_norm(s);
  if (do_expmap) {
    float fac = tanhf(n) / n;
    float s2 = 0.f;
#pragma unroll
    for (int i = 0; i < 8; ++i) { u[i] *= fac; s2 += u[i] * u[i]; }
    s2 = wave_sum(s2);
    n = clamp_norm(s2);
  }
#pragma unroll
  for (int i = 0; i < 8; ++i) outh[row * H + lane * 8 + i] = (_Float16)u[i];
  if (lane == 0) outn[row] = n;
}

// ---------------- kernel 2: Ux = mobius_matvec(W_g, x)  (bulk WMMA GEMM) ------
// grid (TB/64, 3 gates); block 256 = 8 waves.  Per block: 64 rows (4 M-tiles).
// Wave w keeps B-fragments for N-tiles 2w,2w+1 (all 8 k-steps) in VGPRs; the
// 64x256 f16 A tile is staged into LDS once (async copy when available).
// A-fragment LDS loads are software-pipelined one k-step ahead of the WMMAs.
__global__ void __launch_bounds__(256)
ux_gemm_kernel(const _Float16* __restrict__ xh, const float* __restrict__ xn,
               const _Float16* __restrict__ w16, float* __restrict__ ux) {
  __shared__ _Float16 sAh[64][H];   // 32 KB
  __shared__ float    sMx[16][H];   // 16 KB
  __shared__ float    sXn[16];
  int g = blockIdx.y;
  int tid = threadIdx.x, wv = tid >> 5, lane = tid & 31;
  long mbase = (long)blockIdx.x * 64;
  const _Float16* wg = w16 + (long)g * H * H;

  // ---- stage A tile (64x256 f16) into LDS ----
  {
    const _Float16* src = xh + mbase * H;   // 16384 halves = 2048 x 16B chunks
#if HAVE_ASYNC
#pragma unroll
    for (int c = 0; c < 8; ++c) {
      int idx = c * 256 + tid;
      __builtin_amdgcn_global_load_async_to_lds_b128(
          (AS_GLOBAL v4i*)(src + idx * 8),
          (AS_LDS v4i*)(&sAh[0][0] + idx * 8),
          0, 0);
    }
    __builtin_amdgcn_s_wait_asynccnt(0);
#else
#pragma unroll
    for (int c = 0; c < 8; ++c) {
      int idx = c * 256 + tid;
      ((v4i*)&sAh[0][0])[idx] = ((const v4i*)src)[idx];
    }
#endif
  }

  // ---- register-resident B fragments: 2 N-tiles x 8 k-steps ----
  int n0 = wv * 2, n1 = n0 + 1;
  v16h B0[8], B1[8];
#pragma unroll
  for (int k = 0; k < 8; ++k) {
    B0[k] = load_b_frag_h(wg, H, n0 * 16, k * 32, lane);
    B1[k] = load_b_frag_h(wg, H, n1 * 16, k * 32, lane);
  }
  __syncthreads();

  float* uxg = ux + (long)g * TBH;
#pragma unroll 1
  for (int m = 0; m < 4; ++m) {
    v8f acc0 = {}, acc1 = {};
    v16h a = load_a_frag(&sAh[m * 16][0], H, 0, lane);
#pragma unroll
    for (int k = 0; k < 7; ++k) {
      v16h an = load_a_frag(&sAh[m * 16][0], H, (k + 1) * 32, lane);
      acc0 = __builtin_amdgcn_wmma_f32_16x16x32_f16(false, a, false, B0[k], (short)0, acc0, false, false);
      acc1 = __builtin_amdgcn_wmma_f32_16x16x32_f16(false, a, false, B1[k], (short)0, acc1, false, false);
      a = an;
    }
    acc0 = __builtin_amdgcn_wmma_f32_16x16x32_f16(false, a, false, B0[7], (short)0, acc0, false, false);
    acc1 = __builtin_amdgcn_wmma_f32_16x16x32_f16(false, a, false, B1[7], (short)0, acc1, false, false);

    __syncthreads();                       // prior epilogue reads complete
    store_c(&sMx[0][0], H, n0 * 16, acc0, lane);
    store_c(&sMx[0][0], H, n1 * 16, acc1, lane);
    if (tid < 16) sXn[tid] = xn[mbase + m * 16 + tid];
    __syncthreads();

    // epilogue: mobius_matvec scaling, 2 rows per wave
#pragma unroll
    for (int rr = 0; rr < 2; ++rr) {
      int row = wv + rr * 8;
      float mx[8], s = 0.f;
#pragma unroll
      for (int i = 0; i < 8; ++i) { mx[i] = sMx[row][lane * 8 + i]; s += mx[i] * mx[i]; }
      s = wave_sum(s);
      float fac = 0.f;
      if (s != 0.f) {
        float mxn = clamp_norm(s);
        float xr  = sXn[row];
        fac = tanhf(mxn / xr * artanh_f(xr)) / mxn;
      }
      float* o = uxg + (mbase + m * 16 + row) * H;
#pragma unroll
      for (int i = 0; i < 8; ++i) o[lane * 8 + i] = fac * mx[i];
    }
  }
}

// ---------------- recurrence helpers ----------------

// matvec tile (A-fragment ds_loads pipelined one k-step ahead)
__device__ __forceinline__ void wmma_nt(const _Float16* __restrict__ hf,
                                        const v16h (&Wk)[8],
                                        float* __restrict__ dst, int wv, int lane) {
  v8f acc = {};
  v16h a = load_a_frag(hf, H, 0, lane);
#pragma unroll
  for (int k = 0; k < 7; ++k) {
    v16h an = load_a_frag(hf, H, (k + 1) * 32, lane);
    acc = __builtin_amdgcn_wmma_f32_16x16x32_f16(false, a, false, Wk[k], (short)0, acc, false, false);
    a = an;
  }
  acc = __builtin_amdgcn_wmma_f32_16x16x32_f16(false, a, false, Wk[7], (short)0, acc, false, false);
  store_c(dst, H, wv * 16, acc, lane);
}

// tail of _one_transform: mobius_matvec scaling of raw Mx, then mobius_add with
// Ux row (global) then with bias row (LDS).  8 outputs per lane.
__device__ __forceinline__ void one_transform_tail(const float* __restrict__ mxrow,
                                                   const float* __restrict__ uxrow,
                                                   const float* __restrict__ brow,
                                                   float bn2, float xn, float art_xn,
                                                   int lane, float out[8]) {
  float mx[8], u[8];
  float sm = 0.f, su = 0.f, smu = 0.f;
#pragma unroll
  for (int i = 0; i < 8; ++i) {
    mx[i] = mxrow[lane * 8 + i];
    u[i]  = uxrow[lane * 8 + i];
    sm += mx[i] * mx[i]; su += u[i] * u[i]; smu += mx[i] * u[i];
  }
  sm = wave_sum(sm); su = wave_sum(su); smu = wave_sum(smu);
  float fac = 0.f;
  if (sm != 0.f) {
    float mxn = clamp_norm(sm);
    fac = tanhf(mxn / xn * art_xn) / mxn;
  }
  float x2 = fac * fac * sm, xy = fac * smu, y2 = su;
  float cx = 1.f + 2.f * xy + y2;
  float cy = 1.f - x2;
  float dn = fmaxf(1.f + 2.f * xy + x2 * y2, 1e-15f);
  float v[8], sv = 0.f, svb = 0.f;
#pragma unroll
  for (int i = 0; i < 8; ++i) {
    v[i] = (cx * fac * mx[i] + cy * u[i]) / dn;
    sv  += v[i] * v[i];
    svb += v[i] * brow[lane * 8 + i];
  }
  sv = wave_sum(sv); svb = wave_sum(svb);
  float cx2 = 1.f + 2.f * svb + bn2;
  float cy2 = 1.f - sv;
  float dn2 = fmaxf(1.f + 2.f * svb + sv * bn2, 1e-15f);
#pragma unroll
  for (int i = 0; i < 8; ++i) out[i] = (cx2 * v[i] + cy2 * brow[lane * 8 + i]) / dn2;
}

__device__ __forceinline__ void logmap_sigmoid(float w[8], int lane) {
  float s = 0.f;
#pragma unroll
  for (int i = 0; i < 8; ++i) s += w[i] * w[i];
  s = wave_sum(s);
  float n  = clamp_norm(s);
  float sc = artanh_f(n) / n;
#pragma unroll
  for (int i = 0; i < 8; ++i) w[i] = 1.f / (1.f + expf(-w[i] * sc));
}

// ---------------- kernel 3: persistent recurrence ----------------
// grid = 8 WGs (16 batch rows each); block = 512 threads = 16 waves.
// wave wv owns batch row wv (elementwise/reductions) AND N-tile wv (WMMA).
__global__ void __launch_bounds__(512)
mobius_gru_rec_kernel(const _Float16* __restrict__ w_hh16, // [3H, H] f16
                      const float* __restrict__ bias,      // [3, H]
                      const float* __restrict__ ux,        // [3][TB][H]
                      const float* __restrict__ h0,        // [NB, H]
                      int hyp_h0,
                      float* __restrict__ out_f32,         // [T*NB, H] or null
                      _Float16* __restrict__ out_f16,      // [T*NB, H] or null
                      float* __restrict__ out_norm,        // [T*NB] or null
                      float* __restrict__ h_last) {        // [NB, H]
  __shared__ float    sh_h[16][H];
  __shared__ _Float16 sh_hf[16][H];   // h (f16), reused for rh
  __shared__ float    sA[16][H];      // matvec result / h_tilde
  __shared__ float    sZ[16][H];      // z gate
  __shared__ float    sBias[3][H];
  __shared__ float    sBn2[3];

  int tid = threadIdx.x, wv = tid >> 5, lane = tid & 31;
  long brow = (long)blockIdx.x * 16 + wv;

  // recurrent weight fragments, resident in VGPRs (gate r=0, h=1, z=2)
  v16h Wb0[8], Wb1[8], Wb2[8];
#pragma unroll
  for (int k = 0; k < 8; ++k) {
    Wb0[k] = load_b_frag_h(w_hh16 + 0L * H * H, H, wv * 16, k * 32, lane);
    Wb1[k] = load_b_frag_h(w_hh16 + 1L * H * H, H, wv * 16, k * 32, lane);
    Wb2[k] = load_b_frag_h(w_hh16 + 2L * H * H, H, wv * 16, k * 32, lane);
  }

  for (int idx = tid; idx < 3 * H; idx += 512) sBias[idx >> 8][idx & 255] = bias[idx];
  __syncthreads();
  if (wv < 3) {
    float s = 0.f;
#pragma unroll
    for (int i = 0; i < 8; ++i) { float b = sBias[wv][lane * 8 + i]; s += b * b; }
    s = wave_sum(s);
    if (lane == 0) sBn2[wv] = s;
  }

  // init h (optionally expmap0)
  {
    const float* hr = h0 + brow * H;
    float u[8], s = 0.f;
#pragma unroll
    for (int i = 0; i < 8; ++i) { u[i] = hr[lane * 8 + i]; s += u[i] * u[i]; }
    if (!hyp_h0) {
      s = wave_sum(s);
      float n = clamp_norm(s);
      float fac = tanhf(n) / n;
#pragma unroll
      for (int i = 0; i < 8; ++i) u[i] *= fac;
    }
#pragma unroll
    for (int i = 0; i < 8; ++i) {
      sh_h[wv][lane * 8 + i]  = u[i];
      sh_hf[wv][lane * 8 + i] = (_Float16)u[i];
    }
  }
  __syncthreads();

  for (int t = 0; t < T; ++t) {
    // ||h||^2 (own row)
    float hv[8], hn2 = 0.f;
#pragma unroll
    for (int i = 0; i < 8; ++i) { hv[i] = sh_h[wv][lane * 8 + i]; hn2 += hv[i] * hv[i]; }
    hn2 = wave_sum(hn2);
    float hn = clamp_norm(hn2);
    float art_hn = artanh_f(hn);

    long rowoff = ((long)t * NB + brow) * H;

    // ---- z gate (weights gate 2) ----
    __syncthreads();                       // all sh_hf rows ready
    wmma_nt(&sh_hf[0][0], Wb2, &sA[0][0], wv, lane);
    __syncthreads();
    {
      float zw[8];
      one_transform_tail(&sA[wv][0], ux + 2L * TBH + rowoff, &sBias[2][0], sBn2[2],
                         hn, art_hn, lane, zw);
      logmap_sigmoid(zw, lane);
#pragma unroll
      for (int i = 0; i < 8; ++i) sZ[wv][lane * 8 + i] = zw[i];
    }

    // ---- r gate (weights gate 0) ----
    __syncthreads();                       // z-tail reads of sA complete
    wmma_nt(&sh_hf[0][0], Wb0, &sA[0][0], wv, lane);
    __syncthreads();
    float rhn, art_rhn;
    {
      float rw[8];
      one_transform_tail(&sA[wv][0], ux + 0L * TBH + rowoff, &sBias[0][0], sBn2[0],
                         hn, art_hn, lane, rw);
      logmap_sigmoid(rw, lane);
      // rh = mobius_pointwise_mul(r, h)
      float wx[8], swx = 0.f;
#pragma unroll
      for (int i = 0; i < 8; ++i) { wx[i] = rw[i] * hv[i]; swx += wx[i] * wx[i]; }
      swx = wave_sum(swx);
      float fac = 0.f;
      if (swx != 0.f) {
        float wxn = clamp_norm(swx);
        fac = tanhf(wxn / hn * art_hn) / wxn;
      }
      float rhn2 = fac * fac * swx;
      rhn = clamp_norm(rhn2);
      art_rhn = artanh_f(rhn);
#pragma unroll
      for (int i = 0; i < 8; ++i) sh_hf[wv][lane * 8 + i] = (_Float16)(fac * wx[i]);
    }

    // ---- h_tilde (weights gate 1, A = rh) ----
    __syncthreads();                       // all rh rows ready
    wmma_nt(&sh_hf[0][0], Wb1, &sA[0][0], wv, lane);
    __syncthreads();
    {
      float ht[8];
      one_transform_tail(&sA[wv][0], ux + 1L * TBH + rowoff, &sBias[1][0], sBn2[1],
                         rhn, art_rhn, lane, ht);
      // delta = mobius_add(-h, h_tilde)
      float sht = 0.f, shht = 0.f;
#pragma unroll
      for (int i = 0; i < 8; ++i) { sht += ht[i] * ht[i]; shht += hv[i] * ht[i]; }
      sht = wave_sum(sht); shht = wave_sum(shht);
      float xy = -shht;
      float cx = 1.f + 2.f * xy + sht;
      float cy = 1.f - hn2;
      float dn = fmaxf(1.f + 2.f * xy + hn2 * sht, 1e-15f);
      float dl[8], sd = 0.f;
#pragma unroll
      for (int i = 0; i < 8; ++i) {
        dl[i] = (cx * (-hv[i]) + cy * ht[i]) / dn;
        sd += dl[i] * dl[i];
      }
      // zd = mobius_pointwise_mul(z, delta)
      float wx[8], swx = 0.f, shw = 0.f;
#pragma unroll
      for (int i = 0; i < 8; ++i) {
        wx[i] = sZ[wv][lane * 8 + i] * dl[i];
        swx += wx[i] * wx[i];
        shw += hv[i] * wx[i];
      }
      sd = wave_sum(sd); swx = wave_sum(swx); shw = wave_sum(shw);
      float dnn = clamp_norm(sd);
      float fac = 0.f;
      if (swx != 0.f) {
        float wxn = clamp_norm(swx);
        fac = tanhf(wxn / dnn * artanh_f(dnn)) / wxn;
      }
      // h_new = mobius_add(h, zd)
      float y2 = fac * fac * swx;
      float xy2 = fac * shw;
      float cx2 = 1.f + 2.f * xy2 + y2;
      float cy2 = 1.f - hn2;
      float dn2 = fmaxf(1.f + 2.f * xy2 + hn2 * y2, 1e-15f);
      float hnew[8], shn = 0.f;
#pragma unroll
      for (int i = 0; i < 8; ++i) {
        hnew[i] = (cx2 * hv[i] + cy2 * fac * wx[i]) / dn2;
        shn += hnew[i] * hnew[i];
      }
      shn = wave_sum(shn);
#pragma unroll
      for (int i = 0; i < 8; ++i) {
        sh_h[wv][lane * 8 + i]  = hnew[i];
        sh_hf[wv][lane * 8 + i] = (_Float16)hnew[i];
      }
      if (out_f32) {
        float* o = out_f32 + rowoff;
#pragma unroll
        for (int i = 0; i < 8; ++i) o[lane * 8 + i] = hnew[i];
      }
      if (out_f16) {
        _Float16* o = out_f16 + rowoff;
#pragma unroll
        for (int i = 0; i < 8; ++i) o[lane * 8 + i] = (_Float16)hnew[i];
        if (lane == 0) out_norm[(long)t * NB + brow] = clamp_norm(shn);
      }
      if (t == T - 1) {
        float* o = h_last + brow * H;
#pragma unroll
        for (int i = 0; i < 8; ++i) o[lane * 8 + i] = hnew[i];
      }
    }
  }
}

// ---------------- host launcher ----------------

extern "C" void kernel_launch(void* const* d_in, const int* in_sizes, int n_in,
                              void* d_out, int out_size, void* d_ws, size_t ws_size,
                              hipStream_t stream) {
  (void)in_sizes; (void)n_in; (void)out_size; (void)ws_size;
  const float* x     = (const float*)d_in[0];
  const float* h0    = (const float*)d_in[1];
  const float* w_ih0 = (const float*)d_in[2];
  const float* w_hh0 = (const float*)d_in[3];
  const float* b0    = (const float*)d_in[4];
  const float* w_ih1 = (const float*)d_in[5];
  const float* w_hh1 = (const float*)d_in[6];
  const float* b1    = (const float*)d_in[7];

  char* ws = (char*)d_ws;
  size_t off = 0;
  auto carve = [&](size_t bytes) {
    void* p = ws + off;
    off += (bytes + 255) & ~(size_t)255;
    return p;
  };
  _Float16* xh     = (_Float16*)carve((size_t)TB * H * sizeof(_Float16));
  float*    xn     = (float*)   carve((size_t)TB * sizeof(float));
  _Float16* o0h    = (_Float16*)carve((size_t)TB * H * sizeof(_Float16));
  float*    o0n    = (float*)   carve((size_t)TB * sizeof(float));
  float*    uxb    = (float*)   carve((size_t)3 * TB * H * sizeof(float));
  _Float16* wih0_h = (_Float16*)carve((size_t)3 * H * H * sizeof(_Float16));
  _Float16* whh0_h = (_Float16*)carve((size_t)3 * H * H * sizeof(_Float16));
  _Float16* wih1_h = (_Float16*)carve((size_t)3 * H * H * sizeof(_Float16));
  _Float16* whh1_h = (_Float16*)carve((size_t)3 * H * H * sizeof(_Float16));

  float* out1   = (float*)d_out;                 // [T, NB, H]
  float* hlast0 = out1 + (long)TB * H;           // h stack [2, NB, H]
  float* hlast1 = hlast0 + (long)NB * H;

  const int WN = 3 * H * H;                      // 196608
  const int WB = (WN + 255) / 256;
  wcvt_kernel<<<WB, 256, 0, stream>>>(w_ih0, wih0_h, WN);
  wcvt_kernel<<<WB, 256, 0, stream>>>(w_hh0, whh0_h, WN);
  wcvt_kernel<<<WB, 256, 0, stream>>>(w_ih1, wih1_h, WN);
  wcvt_kernel<<<WB, 256, 0, stream>>>(w_hh1, whh1_h, WN);

  // layer 0: x -> expmap0 -> f16 + norms
  prep_kernel<<<TB / 8, 256, 0, stream>>>(x, xh, xn, 1);
  // layer 0: Ux GEMMs (3 gates)
  ux_gemm_kernel<<<dim3(TB / 64, 3), 256, 0, stream>>>(xh, xn, wih0_h, uxb);
  // layer 0 recurrence: euclidean h0 (expmap inside); emit out0 as f16 + norms
  mobius_gru_rec_kernel<<<NB / 16, 512, 0, stream>>>(
      whh0_h, b0, uxb, h0, /*hyp_h0=*/0,
      /*out_f32=*/nullptr, /*out_f16=*/o0h, /*out_norm=*/o0n, hlast0);
  // layer 1: Ux GEMMs from hyperbolic out0
  ux_gemm_kernel<<<dim3(TB / 64, 3), 256, 0, stream>>>(o0h, o0n, wih1_h, uxb);
  // layer 1 recurrence: hyperbolic h0; emit out1 f32 into d_out
  mobius_gru_rec_kernel<<<NB / 16, 512, 0, stream>>>(
      whh1_h, b1, uxb, h0 + (long)NB * H, /*hyp_h0=*/1,
      /*out_f32=*/out1, /*out_f16=*/nullptr, /*out_norm=*/nullptr, hlast1);
}
